// FBPINNWithWindow_30940944401058
// MI455X (gfx1250) — matrix operand order repaired and checked by reading, not compiled, not run on hardware
//
#include <hip/hip_runtime.h>
#include <hip/hip_bf16.h>
#include <math.h>

// ---------------------------------------------------------------------------
// FBPINN forward on gfx1250 (MI455X). Compute-bound (~2.9e11 FLOP vs ~13MB
// traffic) -> all wide GEMMs run on v_wmma_f32_16x16x32_f16 (f16 in, f32 acc).
// Subnet weight tiles are staged into LDS with global_load_async_to_lds_b128
// (ASYNCcnt, double-buffered) so all 4 waves of a block share one copy.
// ---------------------------------------------------------------------------

typedef __attribute__((ext_vector_type(16))) _Float16 v16h;
typedef __attribute__((ext_vector_type(8)))  float    v8f;

#define NPTS 65536
#define NC   16
#define PH   128
#define SW   256

// ---- workspace layout (bytes, all 256-aligned) ----
#define WGT_OFF      0u                      // [N][16] f32      = 4 MB
#define U_OFF        4194304u                // [16][N] f32      = 4 MB
#define POU_MID_OFF  8388608u                // 2 x 128x128 f16  = 64 KB
#define POU_LAST_OFF 8454144u                // 128x16 f16       = 4 KB
#define SUB_MID_OFF  8458240u                // 32 x 256x256 f16 = 4 MB
// total ~12.1 MB

// B-fragment packing for V_WMMA_F32_16X16X32_F16 (B is K=32 x N=16).
// lane = (n%16) + (k>=16 ? 16 : 0); within lane, VGPR j holds K = 2j,2j+1.
// Fragments stored lane-major: 16 contiguous halves per lane.
// kt-major tile order (PoU path):
__device__ __forceinline__ int frag_index(int k, int n, int NT) {
  int kt = k >> 5, kl = k & 31;
  int nt = n >> 4, nl = n & 15;
  int lane = nl + (kl & 16);
  int j = (kl & 15) >> 1;
  int h = kl & 1;
  return ((kt * NT + nt) * 32 + lane) * 16 + j * 2 + h;
}
// nt-major tile order (subnet path; makes a 4-ntile chunk contiguous):
__device__ __forceinline__ int frag_index_nt(int k, int n, int KT) {
  int kt = k >> 5, kl = k & 31;
  int nt = n >> 4, nl = n & 15;
  int lane = nl + (kl & 16);
  int j = (kl & 15) >> 1;
  int h = kl & 1;
  return ((nt * KT + kt) * 32 + lane) * 16 + j * 2 + h;
}

// A-fragment K index, contiguous order (k = 16*hi + s) matching B packing.
__device__ __forceinline__ int a_kidx(int j, int hi) {
  return 2 * j + hi * 16;
}

// ---------------------------------------------------------------------------
// Kernel 1: pack weights f32 -> f16 WMMA B-fragments.
// ---------------------------------------------------------------------------
#define SUBMID_ELEMS (16 * 2 * 256 * 256)   // 2097152
#define POUMID_ELEMS (2 * 128 * 128)        // 32768
#define POULAST_ELEMS (128 * 16)            // 2048
#define PREP_TOTAL (SUBMID_ELEMS + POUMID_ELEMS + POULAST_ELEMS)

__global__ __launch_bounds__(256) void prep_kernel(
    const float* __restrict__ pou_Wmid, const float* __restrict__ pou_Wl,
    const float* __restrict__ sub_Wmid,
    _Float16* __restrict__ pou_mid, _Float16* __restrict__ pou_last,
    _Float16* __restrict__ sub_mid) {
  int idx = blockIdx.x * 256 + threadIdx.x;
  if (idx < SUBMID_ELEMS) {
    int cl = idx >> 16;                 // (c*2+l)
    int rem = idx & 65535;
    int k = rem >> 8, n = rem & 255;
    sub_mid[cl * 65536 + frag_index_nt(k, n, 8)] = (_Float16)sub_Wmid[idx];
  } else if (idx < SUBMID_ELEMS + POUMID_ELEMS) {
    int i = idx - SUBMID_ELEMS;
    int l = i >> 14;
    int rem = i & 16383;
    int k = rem >> 7, n = rem & 127;
    pou_mid[l * 16384 + frag_index(k, n, 8)] = (_Float16)pou_Wmid[i];
  } else if (idx < PREP_TOTAL) {
    int i = idx - SUBMID_ELEMS - POUMID_ELEMS;
    int k = i >> 4, n = i & 15;
    pou_last[frag_index(k, n, 1)] = (_Float16)pou_Wl[i];
  }
}

// ---------------------------------------------------------------------------
// Kernel 2: PoU gating net (~2% of FLOPs). 4 waves/block, 16 rows/wave.
// ---------------------------------------------------------------------------
__global__ __launch_bounds__(128) void pou_kernel(
    const float* __restrict__ x, const float* __restrict__ W0,
    const float* __restrict__ b0, const float* __restrict__ bmid,
    const float* __restrict__ bl, const _Float16* __restrict__ midw,
    const _Float16* __restrict__ lastw, float* __restrict__ wgt) {
  __shared__ _Float16 hbuf[4][16][PH + 8];
  __shared__ float xbuf[4][16][2];
  const int tid = threadIdx.x;
  const int wave = tid >> 5, lane = tid & 31;
  const int hi = lane >> 4, lm = lane & 15;
  const int rowbase = blockIdx.x * 64 + wave * 16;

  if (lane < 16) {
    xbuf[wave][lane][0] = x[(rowbase + lane) * 2 + 0];
    xbuf[wave][lane][1] = x[(rowbase + lane) * 2 + 1];
  }
  __syncthreads();

  // layer 0 (K=2): scalar, relu
  for (int t = 0; t < 4; ++t) {
    int col = lane + 32 * t;
    float w0 = W0[col], w1 = W0[PH + col], bb = b0[col];
    for (int r = 0; r < 16; ++r) {
      float v = xbuf[wave][r][0] * w0 + xbuf[wave][r][1] * w1 + bb;
      hbuf[wave][r][col] = (_Float16)(v > 0.f ? v : 0.f);
    }
  }
  __syncthreads();

  // residual mid layers (WMMA, KT=4 NT=8)
  for (int l = 0; l < 2; ++l) {
    v16h A[4];
#pragma unroll
    for (int kt = 0; kt < 4; ++kt) {
      union { v16h v; unsigned u[8]; } au;
#pragma unroll
      for (int j = 0; j < 8; ++j)
        au.u[j] = *(const unsigned*)&hbuf[wave][lm][kt * 32 + a_kidx(j, hi)];
      A[kt] = au.v;
    }
    const v16h* B = (const v16h*)(midw + l * 16384);
    __syncthreads();
    for (int nt = 0; nt < 8; ++nt) {
      v8f acc = {};
#pragma unroll
      for (int kt = 0; kt < 4; ++kt) {
        v16h b = B[(kt * 8 + nt) * 32 + lane];
        acc = __builtin_amdgcn_wmma_f32_16x16x32_f16(false, A[kt], false, b,
                                                     (short)0, acc, false, false);
      }
      int col = nt * 16 + lm;
      float bb = bmid[l * PH + col];
#pragma unroll
      for (int i = 0; i < 8; ++i) {
        int m = i + hi * 8;
        float v = acc[i] + bb;
        v = v > 0.f ? v : 0.f;
        float prev = (float)hbuf[wave][m][col];
        hbuf[wave][m][col] = (_Float16)(prev + v);
      }
    }
    __syncthreads();
  }

  // head: logits[16x16], K=128 (KT=4, NT=1) + softmax over N (16 lanes)
  v8f acc = {};
#pragma unroll
  for (int kt = 0; kt < 4; ++kt) {
    union { v16h v; unsigned u[8]; } au;
#pragma unroll
    for (int j = 0; j < 8; ++j)
      au.u[j] = *(const unsigned*)&hbuf[wave][lm][kt * 32 + a_kidx(j, hi)];
    v16h b = ((const v16h*)lastw)[kt * 32 + lane];
    acc = __builtin_amdgcn_wmma_f32_16x16x32_f16(false, au.v, false, b,
                                                 (short)0, acc, false, false);
  }
  float bb = bl[lm];
#pragma unroll
  for (int i = 0; i < 8; ++i) {
    float v = acc[i] + bb;
    float mx = v;
    for (int s = 1; s < 16; s <<= 1) mx = fmaxf(mx, __shfl_xor(mx, s, 32));
    float e = __expf(v - mx);
    float ss = e;
    for (int s = 1; s < 16; s <<= 1) ss += __shfl_xor(ss, s, 32);
    int n = rowbase + i + hi * 8;
    wgt[n * NC + lm] = e / ss;
  }
}

// ---------------------------------------------------------------------------
// Async staging helpers (gfx1250 async-to-LDS path, tracked by ASYNCcnt).
// Generic->LDS address: LDS aperture maps addr[31:0] to the LDS offset, so a
// 32-bit truncation of the generic shared pointer is the ds address.
// ---------------------------------------------------------------------------
__device__ __forceinline__ void async_copy16(unsigned lds_addr,
                                             const void* gaddr) {
  asm volatile("global_load_async_to_lds_b128 %0, %1, off"
               :: "v"(lds_addr), "v"((unsigned long long)(uintptr_t)gaddr)
               : "memory");
}
__device__ __forceinline__ void wait_async0() {
  asm volatile("s_wait_asynccnt 0x0" ::: "memory");
}

// copy one 32KB chunk (4 ntiles x 8 ktiles of B fragments), 128 threads,
// 16B per lane per instruction -> 16 async ops per thread.
__device__ __forceinline__ void stage_chunk(const _Float16* gsrc,
                                            unsigned lds_base, int tid) {
#pragma unroll
  for (int i = 0; i < 16; ++i) {
    unsigned off = (unsigned)((i * 128 + tid) * 16);
    async_copy16(lds_base + off, (const char*)gsrc + off);
  }
}

// ---------------------------------------------------------------------------
// Kernel 3: batched subnets. grid = (N/64, C). 4 waves/block, 16 rows/wave.
// Two 256x256 tanh layers on WMMA (KT=8, NT=16). B tiles double-buffered in
// LDS via async copies, shared by all 4 waves (4x L2-traffic reduction).
// ---------------------------------------------------------------------------
__global__ __launch_bounds__(128) void sub_kernel(
    const float* __restrict__ x, const float* __restrict__ centers,
    const float* __restrict__ scales, const float* __restrict__ W0,
    const float* __restrict__ b0, const _Float16* __restrict__ midw,
    const float* __restrict__ bmid, const float* __restrict__ Wl,
    const float* __restrict__ blast, float* __restrict__ u) {
  __shared__ _Float16 gbuf[4][16][SW + 8];
  __shared__ __align__(128) _Float16 Bsh[2][16384];   // 2 x 32KB chunks
  __shared__ float xbuf[4][16][2];
  __shared__ float wl_s[SW];
  const int tid = threadIdx.x;
  const int wave = tid >> 5, lane = tid & 31;
  const int hi = lane >> 4, lm = lane & 15;
  const int c = blockIdx.y;
  const int rowbase = blockIdx.x * 64 + wave * 16;
  const unsigned bsh_addr[2] = {
      (unsigned)(uintptr_t)(&Bsh[0][0]),
      (unsigned)(uintptr_t)(&Bsh[1][0])};

  wl_s[tid] = Wl[c * SW + tid];
  wl_s[tid + 128] = Wl[c * SW + tid + 128];
  if (lane < 16) {
    int row = rowbase + lane;
    xbuf[wave][lane][0] = (x[row * 2 + 0] - centers[c * 2 + 0]) / scales[c * 2 + 0];
    xbuf[wave][lane][1] = (x[row * 2 + 1] - centers[c * 2 + 1]) / scales[c * 2 + 1];
  }
  __syncthreads();

  // layer 0 (K=2): scalar, tanh
  for (int t = 0; t < 8; ++t) {
    int col = lane + 32 * t;
    float w0 = W0[c * 512 + col], w1 = W0[c * 512 + SW + col];
    float bb = b0[c * SW + col];
    for (int r = 0; r < 16; ++r) {
      float v = xbuf[wave][r][0] * w0 + xbuf[wave][r][1] * w1 + bb;
      gbuf[wave][r][col] = (_Float16)tanhf(v);
    }
  }
  __syncthreads();

  // mid layers (WMMA, KT=8, NT=16; 4 chunks of 4 ntiles)
  for (int l = 0; l < 2; ++l) {
    const _Float16* Blayer = midw + (c * 2 + l) * 65536;

    // kick off chunk 0 while we gather A fragments from LDS
    stage_chunk(Blayer, bsh_addr[0], tid);

    v16h A[8];
#pragma unroll
    for (int kt = 0; kt < 8; ++kt) {
      union { v16h v; unsigned uu[8]; } au;
#pragma unroll
      for (int j = 0; j < 8; ++j)
        au.uu[j] = *(const unsigned*)&gbuf[wave][lm][kt * 32 + a_kidx(j, hi)];
      A[kt] = au.v;
    }

    for (int ch = 0; ch < 4; ++ch) {
      wait_async0();        // this wave's share of chunk `ch` has landed
      __syncthreads();      // all waves' shares visible
      if (ch < 3) stage_chunk(Blayer + (ch + 1) * 16384,
                              bsh_addr[(ch + 1) & 1], tid);
      const v16h* Bls = (const v16h*)&Bsh[ch & 1][0];
      for (int ntl = 0; ntl < 4; ++ntl) {
        int nt = ch * 4 + ntl;
        v8f acc = {};
#pragma unroll
        for (int kt = 0; kt < 8; ++kt) {
          v16h b = Bls[(ntl * 8 + kt) * 32 + lane];
          acc = __builtin_amdgcn_wmma_f32_16x16x32_f16(false, A[kt], false, b,
                                                       (short)0, acc, false, false);
        }
        int col = nt * 16 + lm;
        float bb = bmid[(c * 2 + l) * SW + col];
#pragma unroll
        for (int i = 0; i < 8; ++i)
          gbuf[wave][i + hi * 8][col] = (_Float16)tanhf(acc[i] + bb);
      }
    }
    __syncthreads();
  }

  // head: u[r] = g[r] . Wl + b.  lane handles row lm, K parity hi.
  float p = 0.f;
  for (int k = hi; k < SW; k += 2)
    p += (float)gbuf[wave][lm][k] * wl_s[k];
  p += __shfl_xor(p, 16, 32);
  if (hi == 0) u[c * NPTS + rowbase + lm] = p + blast[c];
}

// ---------------------------------------------------------------------------
// Kernel 4: windowed sum + Dirichlet bump.
// ---------------------------------------------------------------------------
__global__ __launch_bounds__(256) void combine_kernel(
    const float* __restrict__ x, const float* __restrict__ wgt,
    const float* __restrict__ u, float* __restrict__ out) {
  int n = blockIdx.x * 256 + threadIdx.x;
  float s = 0.f;
#pragma unroll
  for (int c = 0; c < NC; ++c) s += wgt[n * NC + c] * u[c * NPTS + n];
  float x0 = x[n * 2 + 0], x1 = x[n * 2 + 1];
  out[n] = s * (x0 * (1.f - x0)) * (x1 * (1.f - x1));
}

// ---------------------------------------------------------------------------
extern "C" void kernel_launch(void* const* d_in, const int* in_sizes, int n_in,
                              void* d_out, int out_size, void* d_ws, size_t ws_size,
                              hipStream_t stream) {
  const float* x        = (const float*)d_in[0];
  const float* centers  = (const float*)d_in[1];
  const float* scales   = (const float*)d_in[2];
  const float* pou_W0   = (const float*)d_in[3];
  const float* pou_b0   = (const float*)d_in[4];
  const float* pou_Wmid = (const float*)d_in[5];
  const float* pou_bmid = (const float*)d_in[6];
  const float* pou_Wl   = (const float*)d_in[7];
  const float* pou_bl   = (const float*)d_in[8];
  const float* sub_W0   = (const float*)d_in[9];
  const float* sub_b0   = (const float*)d_in[10];
  const float* sub_Wmid = (const float*)d_in[11];
  const float* sub_bmid = (const float*)d_in[12];
  const float* sub_Wl   = (const float*)d_in[13];
  const float* sub_bl   = (const float*)d_in[14];

  char* ws = (char*)d_ws;
  float*    wgt      = (float*)(ws + WGT_OFF);
  float*    uarr     = (float*)(ws + U_OFF);
  _Float16* pou_mid  = (_Float16*)(ws + POU_MID_OFF);
  _Float16* pou_last = (_Float16*)(ws + POU_LAST_OFF);
  _Float16* sub_mid  = (_Float16*)(ws + SUB_MID_OFF);
  float* out = (float*)d_out;

  // 1) repack weights into WMMA B-fragments (L2-resident afterwards)
  prep_kernel<<<(PREP_TOTAL + 255) / 256, 256, 0, stream>>>(
      pou_Wmid, pou_Wl, sub_Wmid, pou_mid, pou_last, sub_mid);

  // 2) PoU gate -> wgt[N][16]
  pou_kernel<<<NPTS / 64, 128, 0, stream>>>(
      x, pou_W0, pou_b0, pou_bmid, pou_bl, pou_mid, pou_last, wgt);

  // 3) subnets -> u[16][N]
  dim3 sgrid(NPTS / 64, NC);
  sub_kernel<<<sgrid, 128, 0, stream>>>(
      x, centers, scales, sub_W0, sub_b0, sub_mid, sub_bmid, sub_Wl, sub_bl, uarr);

  // 4) windowed sum + bump -> out[N]
  combine_kernel<<<NPTS / 256, 256, 0, stream>>>(x, wgt, uarr, out);
}
